// ResGnn_52510270161088
// MI455X (gfx1250) — matrix operand order repaired and checked by reading, not compiled, not run on hardware
//
#include <hip/hip_runtime.h>
#include <hip/hip_bf16.h>

// Problem constants (match reference)
#define NNODES 50000
#define NEDGES 800000
#define EPRIME (NEDGES + NNODES)   // edges + self loops = 850000
#define DIN    128
#define EDGE_DIM 8
#define NHEAD  4
#define NCHAN  16
#define HC     64
#define NEG_SLOPE 0.2f
#define SELF_FILL 0.01f

typedef __attribute__((ext_vector_type(16))) _Float16 v16h;
typedef __attribute__((ext_vector_type(8)))  float    v8f;

// ---------------------------------------------------------------------------
// Self-loop edge embedding: ehself[c] = 0.01 * sum_k We[k*64 + c], both layers
// ---------------------------------------------------------------------------
__global__ void self_eh_kernel(const float* __restrict__ We0,
                               const float* __restrict__ We1,
                               float* __restrict__ ehs0,
                               float* __restrict__ ehs1) {
  int c = threadIdx.x;
  if (c >= HC) return;
  float s0 = 0.f, s1 = 0.f;
  #pragma unroll
  for (int k = 0; k < EDGE_DIM; ++k) {
    s0 += We0[k * HC + c];
    s1 += We1[k * HC + c];
  }
  ehs0[c] = SELF_FILL * s0;
  ehs1[c] = SELF_FILL * s1;
}

// ---------------------------------------------------------------------------
// Per-layer scratch init: AGG=0 (N*64), AMAX=-inf, DEN=0 (N*4)
// ---------------------------------------------------------------------------
__global__ void init_layer_kernel(float* __restrict__ agg,
                                  float* __restrict__ amax,
                                  float* __restrict__ den) {
  int i = blockIdx.x * blockDim.x + threadIdx.x;
  if (i < NNODES * HC) agg[i] = 0.f;
  if (i < NNODES * NHEAD) {
    amax[i] = -__builtin_huge_valf();
    den[i] = 0.f;
  }
}

// ---------------------------------------------------------------------------
// WMMA node transform: Y[N,64] = X[N,K] @ W[K,64] + bias  (K in {64,128})
// 4 waves/block, each wave computes a 16-row x 64-col strip.
// W is staged once per block into LDS in WMMA B-fragment order (f16), so each
// wave's B fetch is a single 32B ds read; A rows load as float4 (K-contiguous).
// ---------------------------------------------------------------------------
__global__ __launch_bounds__(128)
void gemm_node_transform(const float* __restrict__ X,
                         const float* __restrict__ W,
                         const float* __restrict__ bias,
                         float* __restrict__ Y,
                         int K, int numTiles) {
  // Fragment-ordered weights: idx = ((kk32*4 + nt)*32 + lane)*16 + j
  //   lane 0-15 : N = nt*16 + lane,      K = kk32*32 + {0..7, 16..23}[j]
  //   lane 16-31: N = nt*16 + (lane-16), K = kk32*32 + {8..15, 24..31}[j]
  __shared__ _Float16 Wlds[8192];              // up to K=128: K*64 halves
  const int tid = threadIdx.x;
  const int nhalves = (K / 32) * 2048;
  for (int idx = tid; idx < nhalves; idx += 128) {
    const int j    = idx & 15;
    const int lane = (idx >> 4) & 31;
    const int nt   = (idx >> 9) & 3;
    const int kk32 = idx >> 11;
    const int k = kk32 * 32 + (lane >> 4) * 8 + (j & 7) + (j >> 3) * 16;
    const int n = nt * 16 + (lane & 15);
    Wlds[idx] = (_Float16)W[(size_t)k * HC + n];
  }
  __syncthreads();

  const int wave = tid >> 5;
  const int lane = tid & 31;
  const int tile = blockIdx.x * 4 + wave;
  if (tile >= numTiles) return;

  const int m0   = tile * 16;
  const int mrow = m0 + (lane & 15);
  const int hi   = lane >> 4;                  // lane half-group
  const int akb  = hi ? 8 : 0;                 // A fragment K base per ISA layout

  v8f acc[4] = {};

  for (int kk32 = 0; kk32 < K / 32; ++kk32) {
    // A fragment: 16 halves, K = kk + akb + {0..7, 16..23}; float4 loads.
    const float* xrow = X + (size_t)mrow * K + kk32 * 32 + akb;
    const float4 f0 = *(const float4*)(xrow);
    const float4 f1 = *(const float4*)(xrow + 4);
    const float4 f2 = *(const float4*)(xrow + 16);
    const float4 f3 = *(const float4*)(xrow + 20);
    v16h a;
    a[0]  = (_Float16)f0.x; a[1]  = (_Float16)f0.y;
    a[2]  = (_Float16)f0.z; a[3]  = (_Float16)f0.w;
    a[4]  = (_Float16)f1.x; a[5]  = (_Float16)f1.y;
    a[6]  = (_Float16)f1.z; a[7]  = (_Float16)f1.w;
    a[8]  = (_Float16)f2.x; a[9]  = (_Float16)f2.y;
    a[10] = (_Float16)f2.z; a[11] = (_Float16)f2.w;
    a[12] = (_Float16)f3.x; a[13] = (_Float16)f3.y;
    a[14] = (_Float16)f3.z; a[15] = (_Float16)f3.w;

    #pragma unroll
    for (int nt = 0; nt < 4; ++nt) {
      const v16h b = *(const v16h*)&Wlds[(kk32 * 4 + nt) * 512 + lane * 16];
      acc[nt] = __builtin_amdgcn_wmma_f32_16x16x32_f16(
          /*neg_a=*/false, a, /*neg_b=*/false, b,
          /*c_mod=*/(short)0, acc[nt], /*reuse_a=*/false, /*reuse_b=*/false);
    }
  }

  // C/D layout: VGPR r holds M = r + 8*hi, N = (lane&15) + 16*nt
  const int mb = hi ? 8 : 0;
  const int ncol = lane & 15;
  #pragma unroll
  for (int nt = 0; nt < 4; ++nt) {
    const int n = nt * 16 + ncol;
    const float bv = bias[n];
    #pragma unroll
    for (int r = 0; r < 8; ++r) {
      Y[(size_t)(m0 + mb + r) * HC + n] = acc[nt][r] + bv;
    }
  }
}

// ---------------------------------------------------------------------------
// Float atomic max via signed/unsigned integer trick (IEEE ordering)
// ---------------------------------------------------------------------------
__device__ __forceinline__ void atomicMaxFloat(float* addr, float val) {
  if (val >= 0.f) {
    atomicMax((int*)addr, __float_as_int(val));
  } else {
    atomicMin((unsigned int*)addr, __float_as_uint(val));
  }
}

// ---------------------------------------------------------------------------
// Edge pass 1: alpha[e,h] = sum_c leaky(xl[s]+xr[d]+eh) * att[h,c];
// atomic segment-max into AMAX[dst,h].  One thread per (edge, head).
// ---------------------------------------------------------------------------
__global__ void edge_alpha_kernel(const int* __restrict__ src,
                                  const int* __restrict__ dst,
                                  const float* __restrict__ eattr,
                                  const float* __restrict__ We,
                                  const float* __restrict__ att,
                                  const float* __restrict__ ehself,
                                  const float* __restrict__ XL,
                                  const float* __restrict__ XR,
                                  float* __restrict__ ALPHA,
                                  float* __restrict__ AMAX) {
  int t = blockIdx.x * blockDim.x + threadIdx.x;
  if (t >= EPRIME * NHEAD) return;
  const int e = t >> 2;
  const int h = t & 3;

  int s, d;
  const bool selfloop = (e >= NEDGES);
  if (selfloop) { s = d = e - NEDGES; }
  else          { s = src[e]; d = dst[e]; }

  float ea[EDGE_DIM];
  if (!selfloop) {
    const float4* p = (const float4*)(eattr + (size_t)e * EDGE_DIM);
    float4 a0 = p[0], a1 = p[1];
    ea[0] = a0.x; ea[1] = a0.y; ea[2] = a0.z; ea[3] = a0.w;
    ea[4] = a1.x; ea[5] = a1.y; ea[6] = a1.z; ea[7] = a1.w;
  }

  const float* xlp = XL + (size_t)s * HC + h * NCHAN;
  const float* xrp = XR + (size_t)d * HC + h * NCHAN;
  const float* ap  = att + h * NCHAN;
  float alpha = 0.f;
  #pragma unroll
  for (int c = 0; c < NCHAN; ++c) {
    float ehc;
    if (selfloop) {
      ehc = ehself[h * NCHAN + c];
    } else {
      ehc = 0.f;
      #pragma unroll
      for (int k = 0; k < EDGE_DIM; ++k)
        ehc += ea[k] * We[k * HC + h * NCHAN + c];
    }
    float m = xlp[c] + xrp[c] + ehc;
    m = (m > 0.f) ? m : NEG_SLOPE * m;
    alpha += m * ap[c];
  }
  ALPHA[(size_t)e * NHEAD + h] = alpha;
  atomicMaxFloat(&AMAX[(size_t)d * NHEAD + h], alpha);
}

// ---------------------------------------------------------------------------
// Edge pass 2 (fused exp + denom + UNNORMALIZED aggregate):
//   ex = exp(alpha - amax[dst]); DEN[dst,h] += ex;
//   AGG[dst, h*16+c] += ex * XL[src, h*16+c]
// Normalization by DEN happens per-node in finalize (softmax denom is a
// per-(node,head) scalar), saving an entire 850K-edge pass of traffic.
// ---------------------------------------------------------------------------
__global__ void edge_expagg_kernel(const int* __restrict__ src,
                                   const int* __restrict__ dst,
                                   const float* __restrict__ AMAX,
                                   const float* __restrict__ ALPHA,
                                   float* __restrict__ DEN,
                                   const float* __restrict__ XL,
                                   float* __restrict__ AGG) {
  int t = blockIdx.x * blockDim.x + threadIdx.x;
  if (t >= EPRIME * NHEAD) return;
  const int e = t >> 2;
  const int h = t & 3;
  int s, d;
  if (e >= NEDGES) { s = d = e - NEDGES; }
  else             { s = src[e]; d = dst[e]; }

  const float ex = __expf(ALPHA[(size_t)e * NHEAD + h] -
                          AMAX[(size_t)d * NHEAD + h]);
  atomicAdd(&DEN[(size_t)d * NHEAD + h], ex);

  const float* xlp = XL + (size_t)s * HC + h * NCHAN;
  float* ag = AGG + (size_t)d * HC + h * NCHAN;
  #pragma unroll
  for (int c = 0; c < NCHAN; ++c) {
    atomicAdd(&ag[c], ex * xlp[c]);
  }
}

// ---------------------------------------------------------------------------
// Node finalize: v = relu(AGG/den + bias); residual ? H += v : H = v
// ---------------------------------------------------------------------------
__global__ void finalize_kernel(const float* __restrict__ AGG,
                                const float* __restrict__ DEN,
                                const float* __restrict__ bias,
                                float* __restrict__ Hb,
                                int residual) {
  int i = blockIdx.x * blockDim.x + threadIdx.x;
  if (i >= NNODES * HC) return;
  const int n = i >> 6;
  const int h = (i >> 4) & 3;
  float v = AGG[i] / DEN[n * NHEAD + h] + bias[i & (HC - 1)];
  v = v > 0.f ? v : 0.f;
  Hb[i] = residual ? (Hb[i] + v) : v;
}

// ---------------------------------------------------------------------------
// Final linear: out[n] = dot(H[n,:], linW) + linb
// ---------------------------------------------------------------------------
__global__ void final_linear_kernel(const float* __restrict__ Hb,
                                    const float* __restrict__ linW,
                                    const float* __restrict__ linb,
                                    float* __restrict__ out) {
  int n = blockIdx.x * blockDim.x + threadIdx.x;
  if (n >= NNODES) return;
  const float* hp = Hb + (size_t)n * HC;
  float acc = 0.f;
  #pragma unroll
  for (int c = 0; c < HC; ++c) acc += hp[c] * linW[c];
  out[n] = acc + linb[0];
}

// ---------------------------------------------------------------------------
extern "C" void kernel_launch(void* const* d_in, const int* in_sizes, int n_in,
                              void* d_out, int out_size, void* d_ws, size_t ws_size,
                              hipStream_t stream) {
  const float* x     = (const float*)d_in[0];
  const int*   ei    = (const int*)  d_in[1];
  const float* eattr = (const float*)d_in[2];
  const float* Wl0 = (const float*)d_in[3];  const float* bl0 = (const float*)d_in[4];
  const float* Wr0 = (const float*)d_in[5];  const float* br0 = (const float*)d_in[6];
  const float* We0 = (const float*)d_in[7];  const float* att0 = (const float*)d_in[8];
  const float* bias0 = (const float*)d_in[9];
  const float* Wl1 = (const float*)d_in[10]; const float* bl1 = (const float*)d_in[11];
  const float* Wr1 = (const float*)d_in[12]; const float* br1 = (const float*)d_in[13];
  const float* We1 = (const float*)d_in[14]; const float* att1 = (const float*)d_in[15];
  const float* bias1 = (const float*)d_in[16];
  const float* linW = (const float*)d_in[17]; const float* linb = (const float*)d_in[18];

  const int* srcp = ei;
  const int* dstp = ei + NEDGES;

  // Workspace layout (floats)
  float* ws    = (float*)d_ws;
  float* XL    = ws;  ws += (size_t)NNODES * HC;
  float* XR    = ws;  ws += (size_t)NNODES * HC;
  float* AGG   = ws;  ws += (size_t)NNODES * HC;
  float* Hb    = ws;  ws += (size_t)NNODES * HC;
  float* ALPHA = ws;  ws += (size_t)EPRIME * NHEAD;
  float* AMAX  = ws;  ws += (size_t)NNODES * NHEAD;
  float* DEN   = ws;  ws += (size_t)NNODES * NHEAD;
  float* EHS0  = ws;  ws += HC;
  float* EHS1  = ws;  ws += HC;

  const int numTiles = NNODES / 16;              // 3125 exactly
  const dim3 gemmGrid((numTiles + 3) / 4);       // 4 waves/block
  const dim3 gemmBlock(128);
  const int EPH = EPRIME * NHEAD;                // 3,400,000
  const dim3 edgeGrid((EPH + 255) / 256), edgeBlock(256);
  const dim3 nodeGrid((NNODES * HC + 255) / 256), nodeBlock(256);

  // Self-loop edge embeddings (both layers)
  self_eh_kernel<<<1, 64, 0, stream>>>(We0, We1, EHS0, EHS1);

  // ---------------- Layer 0 ----------------
  init_layer_kernel<<<nodeGrid, nodeBlock, 0, stream>>>(AGG, AMAX, DEN);
  gemm_node_transform<<<gemmGrid, gemmBlock, 0, stream>>>(x, Wl0, bl0, XL, DIN, numTiles);
  gemm_node_transform<<<gemmGrid, gemmBlock, 0, stream>>>(x, Wr0, br0, XR, DIN, numTiles);
  edge_alpha_kernel<<<edgeGrid, edgeBlock, 0, stream>>>(srcp, dstp, eattr, We0, att0,
                                                        EHS0, XL, XR, ALPHA, AMAX);
  edge_expagg_kernel<<<edgeGrid, edgeBlock, 0, stream>>>(srcp, dstp, AMAX, ALPHA,
                                                         DEN, XL, AGG);
  finalize_kernel<<<nodeGrid, nodeBlock, 0, stream>>>(AGG, DEN, bias0, Hb, /*residual=*/0);

  // ---------------- Layer 1 ----------------
  init_layer_kernel<<<nodeGrid, nodeBlock, 0, stream>>>(AGG, AMAX, DEN);
  gemm_node_transform<<<gemmGrid, gemmBlock, 0, stream>>>(Hb, Wl1, bl1, XL, HC, numTiles);
  gemm_node_transform<<<gemmGrid, gemmBlock, 0, stream>>>(Hb, Wr1, br1, XR, HC, numTiles);
  edge_alpha_kernel<<<edgeGrid, edgeBlock, 0, stream>>>(srcp, dstp, eattr, We1, att1,
                                                        EHS1, XL, XR, ALPHA, AMAX);
  edge_expagg_kernel<<<edgeGrid, edgeBlock, 0, stream>>>(srcp, dstp, AMAX, ALPHA,
                                                         DEN, XL, AGG);
  finalize_kernel<<<nodeGrid, nodeBlock, 0, stream>>>(AGG, DEN, bias1, Hb, /*residual=*/1);

  // ---------------- Output head ----------------
  final_linear_kernel<<<(NNODES + 255) / 256, 256, 0, stream>>>(Hb, linW, linb,
                                                                (float*)d_out);
}
